// VQVAE_25005299597389
// MI455X (gfx1250) — compile-verified, matrix operand order checked by reading, not run on hardware
//
#include <hip/hip_runtime.h>
#include <hip/hip_bf16.h>
#include <math.h>

// ---------------- problem constants ----------------
#define BSZ   32
#define CIN   128
#define LSEQ  1024
#define DDIM  256
#define KCB   2048
#define OUT0  (BSZ*LSEQ*12)      // out tensor elems
#define OUT1  (BSZ*LSEQ)         // indices elems
#define NTOK  (BSZ*LSEQ)         // 32768 tokens

typedef __attribute__((ext_vector_type(16))) __bf16 v16bf;
typedef __attribute__((ext_vector_type(8)))  float  v8f;

__device__ inline v8f v8f_zero() {
  v8f v;
#pragma unroll
  for (int i = 0; i < 8; ++i) v[i] = 0.0f;
  return v;
}

// Load a 16-element bf16 WMMA fragment from a row-major LDS tile.
// p = &tile[row*ld + k0 + half*8]; p[0..7] -> K=k0+half*8+0..7,
// p[16..23] -> K=k0+16+half*8+0..7 (documented wave32 16-bit layout).
__device__ inline v16bf frag16(const __bf16* __restrict__ p) {
  v16bf f;
#pragma unroll
  for (int i = 0; i < 8; ++i) { f[i] = p[i]; f[i + 8] = p[i + 16]; }
  return f;
}

__device__ inline v8f wmma_bf16(v16bf a, v16bf b, v8f c) {
  return __builtin_amdgcn_wmma_f32_16x16x32_bf16(false, a, false, b,
                                                 (short)0, c, false, false);
}

// Async global->LDS copy of 16 bytes (tracked by ASYNCcnt).
__device__ inline void async_ld_b128(unsigned lds_addr, const void* gaddr) {
  asm volatile("global_load_async_to_lds_b128 %0, %1, off"
               :: "v"(lds_addr), "v"(gaddr) : "memory");
}
__device__ inline void wait_asynccnt0() {
  asm volatile("s_wait_asynccnt 0" ::: "memory");
}

// ============================================================
// Generic 1-D conv as GEMM:
//   Y[b][m][l] = act(bias[m] + sum_{k,t} W[m][k][t]*X[b][k][l+t-pad])
// WG = 256 threads (8 waves). Tile: M=128, N=64 per workgroup.
// ============================================================
__global__ void __launch_bounds__(256)
conv_gemm_kernel(const float* __restrict__ W,    // [Cout][Cin][taps]
                 const float* __restrict__ X,    // [B][Cin][L]
                 const float* __restrict__ bias, // [Cout]
                 float* __restrict__ Y,          // [B][Cout][L]
                 int Cin, int Cout, int taps, int do_relu)
{
  const int l0  = blockIdx.x * 64;
  const int m0  = blockIdx.y * 128;
  const int b   = blockIdx.z;
  const int tid = threadIdx.x;
  const int wave = tid >> 5;
  const int lane = tid & 31;
  const int half = lane >> 4;
  const int ln15 = lane & 15;

  __shared__ __bf16 sA[128 * 32];   // [M][K]
  __shared__ __bf16 sB[64 * 32];    // [N][K]

  const int mw = (wave >> 1) * 32;  // waves arranged 4(M) x 2(N)
  const int nw = (wave & 1) * 32;

  v8f acc[2][2];
#pragma unroll
  for (int mt = 0; mt < 2; ++mt)
#pragma unroll
    for (int nt = 0; nt < 2; ++nt) acc[mt][nt] = v8f_zero();

  const int pad = taps >> 1;
  for (int t = 0; t < taps; ++t) {
    const int shift = t - pad;
    for (int k0 = 0; k0 < Cin; k0 += 32) {
      // stage A (weights) 128x32; consecutive tid -> consecutive k (coalesced for taps=1)
      for (int i = tid; i < 128 * 32; i += 256) {
        int m = i >> 5, k = i & 31;
        sA[i] = (__bf16)W[(size_t)(m0 + m) * Cin * taps + (size_t)(k0 + k) * taps + t];
      }
      // stage B (activations, shifted) [N=64][K=32];
      // consecutive tid -> consecutive n == consecutive l (coalesced)
      for (int i = tid; i < 64 * 32; i += 256) {
        int n = i & 63, k = i >> 6;
        int l = l0 + n + shift;
        float v = (l >= 0 && l < LSEQ)
                    ? X[((size_t)b * Cin + (k0 + k)) * LSEQ + l] : 0.0f;
        sB[n * 32 + k] = (__bf16)v;
      }
      __syncthreads();
#pragma unroll
      for (int mt = 0; mt < 2; ++mt) {
        v16bf a = frag16(&sA[(mw + mt * 16 + ln15) * 32 + half * 8]);
#pragma unroll
        for (int nt = 0; nt < 2; ++nt) {
          v16bf bb = frag16(&sB[(nw + nt * 16 + ln15) * 32 + half * 8]);
          acc[mt][nt] = wmma_bf16(a, bb, acc[mt][nt]);
        }
      }
      __syncthreads();
    }
  }

  // epilogue: C/D layout -> lane half picks M block (r / r+8), ln15 is N
#pragma unroll
  for (int mt = 0; mt < 2; ++mt)
#pragma unroll
    for (int nt = 0; nt < 2; ++nt)
#pragma unroll
      for (int r = 0; r < 8; ++r) {
        int row = m0 + mw + mt * 16 + half * 8 + r;
        int col = l0 + nw + nt * 16 + ln15;
        float v = acc[mt][nt][r] + bias[row];
        if (do_relu) v = fmaxf(v, 0.0f);
        Y[((size_t)b * Cout + row) * LSEQ + col] = v;
      }
}

// ============================================================
// LayerNorm over last axis (row length 1024), in place. One WG per row.
// ============================================================
__global__ void __launch_bounds__(256)
ln_rows_kernel(float* __restrict__ data,
               const float* __restrict__ g,
               const float* __restrict__ bta)
{
  const int row = blockIdx.x;
  float* p = data + (size_t)row * LSEQ;
  const int tid = threadIdx.x;
  __shared__ float red[256];

  float vals[4];
  float s = 0.0f;
#pragma unroll
  for (int i = 0; i < 4; ++i) { vals[i] = p[tid + i * 256]; s += vals[i]; }
  red[tid] = s;
  __syncthreads();
  for (int off = 128; off > 0; off >>= 1) {
    if (tid < off) red[tid] += red[tid + off];
    __syncthreads();
  }
  const float mu = red[0] * (1.0f / LSEQ);
  __syncthreads();

  float s2 = 0.0f;
#pragma unroll
  for (int i = 0; i < 4; ++i) { float d = vals[i] - mu; s2 += d * d; }
  red[tid] = s2;
  __syncthreads();
  for (int off = 128; off > 0; off >>= 1) {
    if (tid < off) red[tid] += red[tid + off];
    __syncthreads();
  }
  const float rstd = rsqrtf(red[0] * (1.0f / LSEQ) + 1e-5f);

#pragma unroll
  for (int i = 0; i < 4; ++i) {
    int l = tid + i * 256;
    p[l] = (vals[i] - mu) * rstd * g[l] + bta[l];
  }
}

// ============================================================
// pack kernels: fp32 -> bf16 staging arrays for the VQ GEMM
// ============================================================
__global__ void __launch_bounds__(256)
pack_z_kernel(const float* __restrict__ z,   // [B][D][L]
              __bf16* __restrict__ zbf)      // [token][D]
{
  size_t gid = (size_t)blockIdx.x * 256 + threadIdx.x;  // NTOK*DDIM
  int token = (int)(gid >> 8), d = (int)(gid & 255);
  int b = token >> 10, l = token & 1023;
  zbf[gid] = (__bf16)z[((size_t)b * DDIM + d) * LSEQ + l];
}

__global__ void __launch_bounds__(256)
pack_cb_kernel(const float* __restrict__ cb, __bf16* __restrict__ cbbf)
{
  size_t gid = (size_t)blockIdx.x * 256 + threadIdx.x;  // KCB*DDIM
  cbbf[gid] = (__bf16)cb[gid];
}

// ============================================================
// codebook row norms + loss accumulator init
// ============================================================
__global__ void __launch_bounds__(256)
cnorm_kernel(const float* __restrict__ cb, float* __restrict__ cn,
             float* __restrict__ lossp)
{
  const int k = blockIdx.x * 256 + threadIdx.x;
  if (k == 0) *lossp = 0.0f;
  if (k < KCB) {
    float s = 0.0f;
    for (int d = 0; d < DDIM; ++d) { float v = cb[(size_t)k * DDIM + d]; s += v * v; }
    cn[k] = s;
  }
}

// ============================================================
// VQ nearest-neighbor: WG = 128 threads (4 waves), 64 tokens / WG.
// dist = ||c||^2 - 2 z.c ; running (min,argmin) per token; WMMA bf16.
// Tile staging uses async global->LDS b128 copies (ASYNCcnt).
// ============================================================
__global__ void __launch_bounds__(128)
vq_kernel(const __bf16* __restrict__ zbf,   // [token][D] bf16
          const __bf16* __restrict__ cbbf,  // [K][D] bf16
          const float* __restrict__ cnorm,  // [K]
          float* __restrict__ idx_f,        // indices (as float) -> d_out
          int* __restrict__ idx_i)          // indices (int) -> ws
{
  const int tok0 = blockIdx.x * 64;
  const int tid  = threadIdx.x;
  const int wave = tid >> 5;
  const int lane = tid & 31;
  const int half = lane >> 4;
  const int ln15 = lane & 15;

  __shared__ __bf16 sZ[64 * 256];   // [tok][d]  (32 KB)
  __shared__ __bf16 sC[64 * 256];   // [code][d] (32 KB)

  const unsigned ldsZ = (unsigned)(uintptr_t)(void*)&sZ[0];
  const unsigned ldsC = (unsigned)(uintptr_t)(void*)&sC[0];

  // async-stage the token tile (contiguous 32 KB of zbf)
  {
    const char* src = (const char*)(zbf + (size_t)tok0 * DDIM);
    for (int i = tid * 16; i < 64 * 256 * 2; i += 128 * 16)
      async_ld_b128(ldsZ + i, src + i);
  }

  const int m0w = wave * 16;
  float runv[8];
  int   runi[8];
#pragma unroll
  for (int r = 0; r < 8; ++r) { runv[r] = 3.4e38f; runi[r] = 0; }

  for (int c0 = 0; c0 < KCB; c0 += 64) {
    if (c0 + 64 < KCB)
      __builtin_prefetch(cbbf + (size_t)(c0 + 64) * DDIM, 0, 1);
    // async-stage the codebook chunk (contiguous 32 KB of cbbf)
    {
      const char* src = (const char*)(cbbf + (size_t)c0 * DDIM);
      for (int i = tid * 16; i < 64 * 256 * 2; i += 128 * 16)
        async_ld_b128(ldsC + i, src + i);
    }
    wait_asynccnt0();
    __syncthreads();

    v8f acc[4];
#pragma unroll
    for (int nt = 0; nt < 4; ++nt) acc[nt] = v8f_zero();

#pragma unroll
    for (int k0 = 0; k0 < 256; k0 += 32) {
      v16bf a = frag16(&sZ[(m0w + ln15) * 256 + k0 + half * 8]);
#pragma unroll
      for (int nt = 0; nt < 4; ++nt) {
        v16bf bb = frag16(&sC[(nt * 16 + ln15) * 256 + k0 + half * 8]);
        acc[nt] = wmma_bf16(a, bb, acc[nt]);
      }
    }

#pragma unroll
    for (int nt = 0; nt < 4; ++nt) {
      int code = c0 + nt * 16 + ln15;
      float cn = cnorm[code];
#pragma unroll
      for (int r = 0; r < 8; ++r) {
        float dvl = cn - 2.0f * acc[nt][r];
        if (dvl < runv[r] || (dvl == runv[r] && code < runi[r])) {
          runv[r] = dvl; runi[r] = code;
        }
      }
    }
    __syncthreads();
  }

  // butterfly min+argmin across the 16 lanes of each half (wave32)
#pragma unroll
  for (int r = 0; r < 8; ++r) {
    float v = runv[r];
    int   ix = runi[r];
    for (int off = 1; off < 16; off <<= 1) {
      float ov = __shfl_xor(v, off, 16);
      int   oi = __shfl_xor(ix, off, 16);
      if (ov < v || (ov == v && oi < ix)) { v = ov; ix = oi; }
    }
    if (ln15 == 0) {
      int token = tok0 + m0w + half * 8 + r;
      idx_i[token] = ix;
      idx_f[token] = (float)ix;
    }
  }
}

// ============================================================
// gather q = codebook[idx], write to [B][D][L], accumulate commit loss
// one block = one token (256 threads = 256 features)
// ============================================================
__global__ void __launch_bounds__(256)
gather_loss_kernel(const float* __restrict__ z, const float* __restrict__ cb,
                   const int* __restrict__ idx, float* __restrict__ qout,
                   float* __restrict__ lossp)
{
  const int token = blockIdx.x;
  const int d = threadIdx.x;
  const int b = token >> 10, l = token & 1023;
  float q  = cb[(size_t)idx[token] * DDIM + d];
  float zz = z[((size_t)b * DDIM + d) * LSEQ + l];
  qout[((size_t)b * DDIM + d) * LSEQ + l] = q;
  float df = q - zz;

  __shared__ float red[256];
  red[d] = df * df;
  __syncthreads();
  for (int off = 128; off > 0; off >>= 1) {
    if (d < off) red[d] += red[d + off];
    __syncthreads();
  }
  if (d == 0) atomicAdd(lossp, red[0]);
}

// ============================================================
// head: 12x128 pointwise conv + tanh + permute to [B][L][12]
// ============================================================
__global__ void __launch_bounds__(256)
head_kernel(const float* __restrict__ h,   // [B][128][L]
            const float* __restrict__ hw,  // [12][128]
            const float* __restrict__ hb,  // [12]
            float* __restrict__ out)       // [B][L][12]
{
  __shared__ float sw[12 * 128];
  for (int i = threadIdx.x; i < 12 * 128; i += 256) sw[i] = hw[i];
  __syncthreads();

  const int token = blockIdx.x * 256 + threadIdx.x;
  const int b = token >> 10, l = token & 1023;
  float acc[12];
#pragma unroll
  for (int j = 0; j < 12; ++j) acc[j] = 0.0f;
  for (int c = 0; c < CIN; ++c) {
    float x = h[((size_t)b * CIN + c) * LSEQ + l];
#pragma unroll
    for (int j = 0; j < 12; ++j) acc[j] += sw[j * 128 + c] * x;
  }
#pragma unroll
  for (int j = 0; j < 12; ++j)
    out[(size_t)token * 12 + j] = tanhf(acc[j] + hb[j]);
}

__global__ void finalize_loss_kernel(const float* __restrict__ lossp,
                                     float* __restrict__ out)
{
  *out = *lossp * (1.0f / (float)((size_t)NTOK * DDIM));
}

// ============================================================
// launch
// ============================================================
extern "C" void kernel_launch(void* const* d_in, const int* in_sizes, int n_in,
                              void* d_out, int out_size, void* d_ws, size_t ws_size,
                              hipStream_t stream)
{
  const float* x      = (const float*)d_in[0];
  const float* enc_w1 = (const float*)d_in[1];
  const float* enc_b1 = (const float*)d_in[2];
  const float* enc_w2 = (const float*)d_in[3];
  const float* enc_b2 = (const float*)d_in[4];
  const float* ln1_g  = (const float*)d_in[5];
  const float* ln1_b  = (const float*)d_in[6];
  const float* cbk    = (const float*)d_in[7];
  const float* dec_w1 = (const float*)d_in[8];
  const float* dec_b1 = (const float*)d_in[9];
  const float* dec_w2 = (const float*)d_in[10];
  const float* dec_b2 = (const float*)d_in[11];
  const float* ln2_g  = (const float*)d_in[12];
  const float* ln2_b  = (const float*)d_in[13];
  const float* head_w = (const float*)d_in[14];
  const float* head_b = (const float*)d_in[15];

  char* ws = (char*)d_ws;
  size_t o = 0;
  float* bufA = (float*)(ws + o); o += (size_t)BSZ * DDIM * LSEQ * 4;  // h1 -> q
  float* bufB = (float*)(ws + o); o += (size_t)BSZ * DDIM * LSEQ * 4;  // h2/z -> hd2
  float* bufC = (float*)(ws + o); o += (size_t)BSZ * CIN  * LSEQ * 4;  // hd1
  __bf16* zbf  = (__bf16*)(ws + o); o += (size_t)NTOK * DDIM * 2;
  __bf16* cbbf = (__bf16*)(ws + o); o += (size_t)KCB * DDIM * 2;
  float* cnorm = (float*)(ws + o); o += (size_t)KCB * 4;
  int*   idxi  = (int*)(ws + o);  o += (size_t)NTOK * 4;
  float* lossp = (float*)(ws + o); o += 64;

  float* out     = (float*)d_out;
  float* idx_f   = out + OUT0;
  float* lossout = out + OUT0 + OUT1;

  // encoder conv1 (k=1, 128->256) + relu
  conv_gemm_kernel<<<dim3(LSEQ / 64, 2, BSZ), 256, 0, stream>>>(
      enc_w1, x, enc_b1, bufA, CIN, DDIM, 1, 1);
  // encoder conv2 (k=3, 256->256) + relu
  conv_gemm_kernel<<<dim3(LSEQ / 64, 2, BSZ), 256, 0, stream>>>(
      enc_w2, bufA, enc_b2, bufB, DDIM, DDIM, 3, 1);
  // LayerNorm over L (rows = B*D)
  ln_rows_kernel<<<BSZ * DDIM, 256, 0, stream>>>(bufB, ln1_g, ln1_b);
  // pack z and codebook to bf16 for the VQ GEMM
  pack_z_kernel<<<(NTOK * DDIM) / 256, 256, 0, stream>>>(bufB, zbf);
  pack_cb_kernel<<<(KCB * DDIM) / 256, 256, 0, stream>>>(cbk, cbbf);
  // codebook norms + loss init
  cnorm_kernel<<<KCB / 256, 256, 0, stream>>>(cbk, cnorm, lossp);
  // VQ argmin
  vq_kernel<<<NTOK / 64, 128, 0, stream>>>(zbf, cbbf, cnorm, idx_f, idxi);
  // gather q + commit loss (q -> bufA, straight-through forward value)
  gather_loss_kernel<<<NTOK, 256, 0, stream>>>(bufB, cbk, idxi, bufA, lossp);
  // decoder conv1 (k=3, 256->128) + relu
  conv_gemm_kernel<<<dim3(LSEQ / 64, 1, BSZ), 256, 0, stream>>>(
      dec_w1, bufA, dec_b1, bufC, DDIM, CIN, 3, 1);
  // decoder conv2 (k=1, 128->128), no relu
  conv_gemm_kernel<<<dim3(LSEQ / 64, 1, BSZ), 256, 0, stream>>>(
      dec_w2, bufC, dec_b2, bufB, CIN, CIN, 1, 0);
  // LayerNorm over L (rows = B*C)
  ln_rows_kernel<<<BSZ * CIN, 256, 0, stream>>>(bufB, ln2_g, ln2_b);
  // head + tanh + permute
  head_kernel<<<NTOK / 256, 256, 0, stream>>>(bufB, head_w, head_b, out);
  // commit loss finalize
  finalize_loss_kernel<<<1, 1, 0, stream>>>(lossp, lossout);

  (void)in_sizes; (void)n_in; (void)out_size; (void)ws_size;
}
